// Attention_5411658793055
// MI455X (gfx1250) — compile-verified, matrix operand order checked by reading, not run on hardware
//
#include <hip/hip_runtime.h>

typedef __attribute__((ext_vector_type(16))) __bf16        v16bf;
typedef __attribute__((ext_vector_type(8)))  float          v8f;
typedef __attribute__((ext_vector_type(4)))  float          v4f;
typedef __attribute__((ext_vector_type(4)))  unsigned int   v4u;
typedef __attribute__((ext_vector_type(8)))  unsigned short v8us;
typedef __attribute__((ext_vector_type(16))) unsigned short v16us;
typedef __attribute__((ext_vector_type(4)))  unsigned int   u32x4;
typedef __attribute__((ext_vector_type(8)))  int            i32x8;
typedef __attribute__((ext_vector_type(4)))  int            i32x4;

#define S_  8192
#define KD_ 512
#define DD_ 512

__device__ __forceinline__ unsigned short f2bf(float f) {
  unsigned int u = __builtin_bit_cast(unsigned int, f);
  u += 0x7FFFu + ((u >> 16) & 1u);           // round-to-nearest-even
  return (unsigned short)(u >> 16);
}

// Load one bf16 fragment slice for this lane (A or B layout, ISA 7.12.2):
// lane-half h holds K elements {8h..8h+7, 16+8h..16+8h+7}, koff = 8*h.
__device__ __forceinline__ v16bf load_frag(const unsigned short* p, int koff) {
  v8us g0 = __builtin_bit_cast(v8us, *reinterpret_cast<const v4u*>(p + koff));
  v8us g1 = __builtin_bit_cast(v8us, *reinterpret_cast<const v4u*>(p + 16 + koff));
  v16us w = __builtin_shufflevector(g0, g1, 0,1,2,3,4,5,6,7,8,9,10,11,12,13,14,15);
  return __builtin_bit_cast(v16bf, w);
}

__device__ __forceinline__ v8f wmma_bf16(v16bf a, v16bf b, v8f c) {
  return __builtin_amdgcn_wmma_f32_16x16x32_bf16(false, a, false, b, (short)0, c,
                                                 false, false);
}

// ---- Tensor Data Mover: issue a 2-D bf16 tile load Global -> LDS ----
// D# per CDNA5 ISA ch.8: group0 {count=1, lds_addr, global_addr, type=2},
// group1 {data_size=2B, tensor_dim0/1, tile_dim0/1, tensor_dim0_stride}.
// 6-arg builtin form (clang-23 / therock-10.0 headers).
__device__ __forceinline__ void tdm_load_2d(const unsigned short* gptr,
                                            unsigned int ldsOff,
                                            unsigned int tensor_d0,
                                            unsigned int tensor_d1,
                                            unsigned int tile_d0,
                                            unsigned int tile_d1,
                                            unsigned int stride0) {
  unsigned long long ga = (unsigned long long)gptr;
  u32x4 g0;
  g0[0] = 1u;                                   // count=1, user mode
  g0[1] = ldsOff;                               // lds_addr (bytes)
  g0[2] = (unsigned int)ga;                     // global_addr[31:0]
  g0[3] = (unsigned int)(ga >> 32) | 0x80000000u; // global_addr[56:32] | type=2<<30
  i32x8 g1;
  g1[0] = (int)(1u << 16);                      // workgroup_mask=0, data_size=1 (2B)
  g1[1] = (int)((tensor_d0 & 0xFFFFu) << 16);   // tensor_dim0[15:0] @ bits63:48
  g1[2] = (int)((tensor_d0 >> 16) | ((tensor_d1 & 0xFFFFu) << 16));
  g1[3] = (int)((tensor_d1 >> 16) | ((tile_d0 & 0xFFFFu) << 16));
  g1[4] = (int)(tile_d1 & 0xFFFFu);             // tile_dim1; tile_dim2=0
  g1[5] = (int)stride0;                         // tensor_dim0_stride[31:0]
  g1[6] = 0;                                    // stride0 hi | stride1 lo
  g1[7] = 0;
  i32x4 gz4 = {0, 0, 0, 0};
  i32x8 gz8 = {0, 0, 0, 0, 0, 0, 0, 0};
  __builtin_amdgcn_tensor_load_to_lds(g0, g1, gz4, gz4, gz8, 0);
}

// ---------------- precision-conversion kernels ----------------
__global__ void cvt_bf16_kernel(const float* __restrict__ src,
                                unsigned short* __restrict__ dst, int n) {
  int i = blockIdx.x * blockDim.x + threadIdx.x;
  if (i < n) dst[i] = f2bf(src[i]);
}

__global__ void cvt_tr_bf16_kernel(const float* __restrict__ src,
                                   unsigned short* __restrict__ dst,
                                   int rows, int cols) {
  int i = blockIdx.x * blockDim.x + threadIdx.x;
  if (i < rows * cols) {
    int r = i / cols, c = i % cols;
    dst[(size_t)c * rows + r] = f2bf(src[i]);   // dst is [cols][rows]
  }
}

// ---------------- flash attention: O = softmax(QK^T/sqrt(d)) V ----------------
__global__ __launch_bounds__(128) void flash_attn_kernel(
    const unsigned short* __restrict__ Qb,   // [S][KD] bf16
    const unsigned short* __restrict__ Kb,   // [S][KD] bf16
    const unsigned short* __restrict__ Vt,   // [KD][S] bf16 (V transposed)
    unsigned short* __restrict__ Ob) {       // [S][DD] bf16 out
  // Double-buffered TDM staging: K tile [32][512], V slice [512][32]
  __shared__ unsigned short ldsK[2][32][512];   // 64 KB
  __shared__ unsigned short ldsV[2][512][32];   // 64 KB
  __shared__ float          ldsP[4][16][32];    //  8 KB

  const int lane = threadIdx.x & 31;
  const int wave = threadIdx.x >> 5;
  const int half = lane >> 4;
  const int l16  = lane & 15;
  const int koff = half * 8;
  const int tile = blockIdx.x * 4 + wave;
  const int row0 = tile * 16;

  // Q tile resident as 16 A-fragments (16x32 each)
  v16bf qa[16];
  #pragma unroll
  for (int ks = 0; ks < 16; ++ks)
    qa[ks] = load_frag(Qb + (size_t)(row0 + l16) * KD_ + ks * 32, koff);

  v8f zero = {};
  v8f o[32];
  #pragma unroll
  for (int t = 0; t < 32; ++t) o[t] = zero;

  float m_[8], l_[8];
  #pragma unroll
  for (int j = 0; j < 8; ++j) { m_[j] = -3.0e38f; l_[j] = 0.0f; }

  const float scale = 0.044194173824159216f;   // 1/sqrt(512)

  const unsigned int kOff0 = (unsigned int)(unsigned long long)&ldsK[0][0][0];
  const unsigned int kOff1 = (unsigned int)(unsigned long long)&ldsK[1][0][0];
  const unsigned int vOff0 = (unsigned int)(unsigned long long)&ldsV[0][0][0];
  const unsigned int vOff1 = (unsigned int)(unsigned long long)&ldsV[1][0][0];

  // prologue: stage tiles for iteration 0 into buffer 0
  if (wave == 0) {
    tdm_load_2d(Kb,  kOff0, KD_, S_, KD_, 32, KD_);   // K rows 0..31
    tdm_load_2d(Vt,  vOff0, S_, KD_, 32, KD_, S_);    // Vt[:, 0:32]
  }

  const int NIT = S_ / 32;
  for (int it = 0; it < NIT; ++it) {
    const int buf = it & 1;
    const int j0  = it * 32;

    if (wave == 0) {
      if (it + 1 < NIT) {
        const int nb = buf ^ 1;
        const int nj = j0 + 32;
        tdm_load_2d(Kb + (size_t)nj * KD_, nb ? kOff1 : kOff0,
                    KD_, S_, KD_, 32, KD_);
        tdm_load_2d(Vt + nj, nb ? vOff1 : vOff0,
                    S_, KD_, 32, KD_, S_);
        __builtin_amdgcn_s_wait_tensorcnt(2);   // oldest 2 (current buf) done
      } else {
        __builtin_amdgcn_s_wait_tensorcnt(0);
      }
    }
    __syncthreads();                            // tile visible to all waves

    // ---- S tiles: s0 = Q*K[j0..j0+15]^T, s1 = Q*K[j0+16..j0+31]^T (from LDS)
    v8f s0 = zero, s1 = zero;
    #pragma unroll
    for (int ks = 0; ks < 16; ++ks) {
      v16bf b0 = load_frag(&ldsK[buf][l16][ks * 32], koff);
      s0 = wmma_bf16(qa[ks], b0, s0);
      v16bf b1 = load_frag(&ldsK[buf][16 + l16][ks * 32], koff);
      s1 = wmma_bf16(qa[ks], b1, s1);
    }

    // ---- online softmax over this 32-key block
    float corr[8];
    #pragma unroll
    for (int j = 0; j < 8; ++j) {
      float a0 = s0[j] * scale;
      float a1 = s1[j] * scale;
      float mx = fmaxf(a0, a1);                 // row reduce across 16 lanes
      mx = fmaxf(mx, __shfl_xor(mx, 1, 32));
      mx = fmaxf(mx, __shfl_xor(mx, 2, 32));
      mx = fmaxf(mx, __shfl_xor(mx, 4, 32));
      mx = fmaxf(mx, __shfl_xor(mx, 8, 32));
      float mn = fmaxf(m_[j], mx);
      float c  = __expf(m_[j] - mn);
      float p0 = __expf(a0 - mn);
      float p1 = __expf(a1 - mn);
      float rs = p0 + p1;
      rs += __shfl_xor(rs, 1, 32);
      rs += __shfl_xor(rs, 2, 32);
      rs += __shfl_xor(rs, 4, 32);
      rs += __shfl_xor(rs, 8, 32);
      l_[j] = l_[j] * c + rs;
      m_[j] = mn;
      corr[j] = c;
      ldsP[wave][j + 8 * half][l16]      = p0;  // C-layout stash for transpose
      ldsP[wave][j + 8 * half][16 + l16] = p1;
    }

    // rescale running O accumulator
    #pragma unroll
    for (int t = 0; t < 32; ++t) {
      #pragma unroll
      for (int j = 0; j < 8; ++j) o[t][j] *= corr[j];
    }

    // ---- P: C-layout -> A-layout via LDS (wave-private region)
    asm volatile("s_wait_dscnt 0" ::: "memory");
    const float* pb = &ldsP[wave][l16][0];
    v4f f0 = *reinterpret_cast<const v4f*>(pb + koff);
    v4f f1 = *reinterpret_cast<const v4f*>(pb + koff + 4);
    v4f f2 = *reinterpret_cast<const v4f*>(pb + 16 + koff);
    v4f f3 = *reinterpret_cast<const v4f*>(pb + 16 + koff + 4);
    v16us pw;
    #pragma unroll
    for (int i = 0; i < 4; ++i) {
      pw[i]      = f2bf(f0[i]);
      pw[4 + i]  = f2bf(f1[i]);
      pw[8 + i]  = f2bf(f2[i]);
      pw[12 + i] = f2bf(f3[i]);
    }
    v16bf pa = __builtin_bit_cast(v16bf, pw);

    // ---- O += P (16x32) @ V (32 x 512): B fragments from staged Vt slice
    #pragma unroll
    for (int t = 0; t < 32; ++t) {
      v16bf vb = load_frag(&ldsV[buf][t * 16 + l16][0], koff);
      o[t] = wmma_bf16(pa, vb, o[t]);
    }

    __syncthreads();   // everyone done reading buf before it is re-staged
  }

  // ---- normalize rows and emit bf16 O for the epilogue GEMM
  float inv[8];
  #pragma unroll
  for (int j = 0; j < 8; ++j) inv[j] = 1.0f / l_[j];
  #pragma unroll
  for (int t = 0; t < 32; ++t) {
    #pragma unroll
    for (int j = 0; j < 8; ++j) {
      int r = row0 + j + 8 * half;
      Ob[(size_t)r * DD_ + t * 16 + l16] = f2bf(o[t][j] * inv[j]);
    }
  }
}

// ---------------- epilogue GEMM: out = O @ Wo ----------------
__global__ __launch_bounds__(128) void gemm_out_kernel(
    const unsigned short* __restrict__ Ob,    // [S][KD] bf16
    const unsigned short* __restrict__ Wot,   // [DD][KD] bf16 (Wo transposed)
    float* __restrict__ out) {                // [S][DD] f32
  const int lane = threadIdx.x & 31;
  const int wave = threadIdx.x >> 5;
  const int half = lane >> 4;
  const int l16  = lane & 15;
  const int koff = half * 8;
  const int mtile = blockIdx.x * 4 + wave;
  const int ntile = blockIdx.y;

  v8f acc = {};
  #pragma unroll
  for (int ks = 0; ks < 16; ++ks) {
    v16bf a = load_frag(Ob  + (size_t)(mtile * 16 + l16) * KD_ + ks * 32, koff);
    v16bf b = load_frag(Wot + (size_t)(ntile * 16 + l16) * KD_ + ks * 32, koff);
    acc = wmma_bf16(a, b, acc);
  }
  #pragma unroll
  for (int j = 0; j < 8; ++j) {
    int r = mtile * 16 + j + 8 * half;
    out[(size_t)r * DD_ + ntile * 16 + l16] = acc[j];
  }
}

// ---------------- host entry ----------------
extern "C" void kernel_launch(void* const* d_in, const int* in_sizes, int n_in,
                              void* d_out, int out_size, void* d_ws, size_t ws_size,
                              hipStream_t stream) {
  const float* Q  = (const float*)d_in[0];
  const float* K  = (const float*)d_in[1];
  const float* V  = (const float*)d_in[2];
  const float* Wo = (const float*)d_in[3];
  float* out = (float*)d_out;

  unsigned short* Qb  = (unsigned short*)d_ws;              // [S][KD]
  unsigned short* Kb  = Qb + (size_t)S_ * KD_;              // [S][KD]
  unsigned short* Vt  = Kb + (size_t)S_ * KD_;              // [KD][S]
  unsigned short* Ob  = Vt + (size_t)S_ * KD_;              // [S][DD]
  unsigned short* Wot = Ob + (size_t)S_ * DD_;              // [DD][KD]

  const int n = S_ * KD_;
  cvt_bf16_kernel<<<n / 256, 256, 0, stream>>>(Q, Qb, n);
  cvt_bf16_kernel<<<n / 256, 256, 0, stream>>>(K, Kb, n);
  cvt_tr_bf16_kernel<<<n / 256, 256, 0, stream>>>(V, Vt, S_, KD_);
  cvt_tr_bf16_kernel<<<(KD_ * DD_) / 256, 256, 0, stream>>>(Wo, Wot, KD_, DD_);

  flash_attn_kernel<<<S_ / 64, 128, 0, stream>>>(Qb, Kb, Vt, Ob);

  dim3 g2(S_ / 64, DD_ / 16);
  gemm_out_kernel<<<g2, 128, 0, stream>>>(Ob, Wot, out);
}
